// HypercubeTransformerGPU_64828236366553
// MI455X (gfx1250) — matrix operand order repaired and checked by reading, not compiled, not run on hardware
//
#include <hip/hip_runtime.h>
#include <math.h>

#define S_LEN 512
#define BATCH 2
#define DM    192
#define DV    32
#define D3    64
#define DFF   768
#define NVOC  256
#define MROWS (BATCH * S_LEN)

typedef __attribute__((ext_vector_type(16))) __bf16 bf16x16;
typedef __attribute__((ext_vector_type(8)))  float  f32x8;
typedef __bf16 dbf16;

// ---------------------------------------------------------------------------
// WMMA fragment loaders (wave32, V_WMMA_F32_16X16X32_BF16; contiguous k order used for both A and B)
// A 16x32 (MxK): lane row M=l&15, K=(l>>4)*16 + 0..15 in element order
// B 32x16 (KxN): lane col N=l&15, K=(l>>4)*16 + 0..15 in element order
// C/D 16x16 f32: lane col N=l&15, rows M=(l>>4)*8 + r in acc[r]
// Both operands are stored so the 16-element runs are contiguous in memory.
// ---------------------------------------------------------------------------
__device__ inline bf16x16 load_a_frag(const dbf16* A, int lda, int mBase, int k0, int lane) {
  int r = lane & 15, h = lane >> 4;
  const dbf16* p = A + (size_t)(mBase + r) * lda + k0 + h * 16;
  bf16x16 f;
#pragma unroll
  for (int i = 0; i < 16; ++i) f[i] = p[i];
  return f;
}
__device__ inline bf16x16 load_b_frag(const dbf16* Bw, int ldb, int nBase, int k0, int lane) {
  // B stored (N,K) row-major -> contiguous in K
  int c = lane & 15, h = lane >> 4;
  const dbf16* p = Bw + (size_t)(nBase + c) * ldb + k0 + h * 16;
  bf16x16 f;
#pragma unroll
  for (int i = 0; i < 16; ++i) f[i] = p[i];
  return f;
}

// ---------------------------------------------------------------------------
// Generic bf16 WMMA GEMM, 32x32 register tile per wave:
//   2 A-frags x 2 B-frags -> 4 independent WMMAs per 32-deep k-step.
// D(M,N) = A(M,K) * B^T [+Cin] [+bias] [gelu]; B stored (N,K) row-major.
// causal clamps K to the lower triangle (A rows zero beyond the diagonal).
// Requires M%32==0, N%32==0, K%32==0 (true for all call sites here).
// ---------------------------------------------------------------------------
__global__ void gemm_wmma_kernel(const dbf16* __restrict__ A, int lda,
                                 const dbf16* __restrict__ B, int ldb,
                                 const float* __restrict__ Cin,
                                 float* __restrict__ D, dbf16* __restrict__ Dbf, int ldd,
                                 const float* __restrict__ bias,
                                 int K, int causal, int act) {
  int lane  = threadIdx.x;
  int mBase = blockIdx.x * 32, nBase = blockIdx.y * 32;
  int c = lane & 15, h = lane >> 4;

  f32x8 acc[2][2];
#pragma unroll
  for (int mi = 0; mi < 2; ++mi)
#pragma unroll
    for (int ni = 0; ni < 2; ++ni)
#pragma unroll
      for (int r = 0; r < 8; ++r)
        acc[mi][ni][r] = Cin
            ? Cin[(size_t)(mBase + mi * 16 + h * 8 + r) * ldd + nBase + ni * 16 + c]
            : 0.f;

  int kend = K;
  if (causal) { int lim = mBase + 32; kend = lim < K ? lim : K; }

  for (int k0 = 0; k0 < kend; k0 += 32) {
    if (k0 + 32 < kend)
      __builtin_prefetch(A + (size_t)mBase * lda + k0 + 32, 0, 3);
    bf16x16 a0 = load_a_frag(A, lda, mBase,      k0, lane);
    bf16x16 a1 = load_a_frag(A, lda, mBase + 16, k0, lane);
    bf16x16 b0 = load_b_frag(B, ldb, nBase,      k0, lane);
    bf16x16 b1 = load_b_frag(B, ldb, nBase + 16, k0, lane);
    acc[0][0] = __builtin_amdgcn_wmma_f32_16x16x32_bf16(false, a0, false, b0,
                                                        (short)0, acc[0][0], false, false);
    acc[0][1] = __builtin_amdgcn_wmma_f32_16x16x32_bf16(false, a0, false, b1,
                                                        (short)0, acc[0][1], false, false);
    acc[1][0] = __builtin_amdgcn_wmma_f32_16x16x32_bf16(false, a1, false, b0,
                                                        (short)0, acc[1][0], false, false);
    acc[1][1] = __builtin_amdgcn_wmma_f32_16x16x32_bf16(false, a1, false, b1,
                                                        (short)0, acc[1][1], false, false);
  }

#pragma unroll
  for (int mi = 0; mi < 2; ++mi)
#pragma unroll
    for (int ni = 0; ni < 2; ++ni) {
      float bv = bias ? bias[nBase + ni * 16 + c] : 0.f;
#pragma unroll
      for (int r = 0; r < 8; ++r) {
        float v = acc[mi][ni][r] + bv;
        if (act == 1) v = 0.5f * v * (1.f + erff(v * 0.70710678118f));  // exact gelu
        size_t idx = (size_t)(mBase + mi * 16 + h * 8 + r) * ldd + nBase + ni * 16 + c;
        D[idx] = v;
        if (Dbf) Dbf[idx] = (dbf16)v;
      }
    }
}

// 32x32 LDS-tiled transpose with f32 -> bf16 convert: dst(C,R) = src(R,C)^T.
// Puts GEMM B-operands into (N,K) layout so all fragment loads are contiguous.
__global__ void transpose_bf16_kernel(const float* __restrict__ src, dbf16* __restrict__ dst,
                                      int R, int C) {
  __shared__ float tile[32][33];
  int rb = blockIdx.x * 32, cb = blockIdx.y * 32;
  int tx = threadIdx.x, ty = threadIdx.y;          // blockDim (32,8)
  for (int yy = ty; yy < 32; yy += 8)
    tile[yy][tx] = src[(size_t)(rb + yy) * C + cb + tx];
  __syncthreads();
  for (int yy = ty; yy < 32; yy += 8)
    dst[(size_t)(cb + yy) * R + rb + tx] = (dbf16)tile[tx][yy];
}

// ---------------------------------------------------------------------------
// Layer prep: fold spectral embedding + adjacency bias into Veff (64x32 bf16),
// gate means + modal masks into wvec (4x64 f32).
// Walsh eigenvectors of Q6 Laplacian: chi_{2^s}(v) = (-1)^{(v>>s)&1} / 8.
// Veff = vert + ADJ_W * (ADJ_BIAS fold), ADJ_BIAS[u,v]=0.5 iff popcount(u^v)==1
// ---------------------------------------------------------------------------
__global__ void prep_layer_kernel(const float* __restrict__ vertex_emb,
                                  const float* __restrict__ Wspec,
                                  const float* __restrict__ vertex_gates,
                                  dbf16* __restrict__ Veff, float* __restrict__ wvec) {
  __shared__ float vert[64 * 32];
  int t = threadIdx.x;                     // blockDim = 512
  for (int idx = t; idx < 2048; idx += 512) {
    int v = idx >> 5, d = idx & 31;
    float sv = vertex_emb[idx];
#pragma unroll
    for (int s = 0; s < 6; ++s) {
      float sp = ((v >> s) & 1) ? -0.125f : 0.125f;
      sv += 0.1f * sp * Wspec[d * 6 + s];  // SPEC_W * spec @ Wspec^T
    }
    vert[idx] = sv;
  }
  __syncthreads();
  for (int idx = t; idx < 2048; idx += 512) {
    int v = idx >> 5, d = idx & 31;
    float acc = vert[idx];
#pragma unroll
    for (int e = 0; e < 6; ++e) acc += 0.05f * vert[((v ^ (1 << e)) << 5) | d];
    Veff[idx] = (dbf16)acc;
  }
  if (t < 64) {
    float s = 0.f;
    for (int cc = 0; cc < DM; ++cc)
      s += 1.f / (1.f + expf(-vertex_gates[t * DM + cc]));
    wvec[t]       = s / (float)DM;                 // gate_mean
    wvec[64 + t]  = (t & 3)  ? 1.f : 0.f;          // AP mask
    wvec[128 + t] = (t & 12) ? 1.f : 0.f;          // BD mask
    wvec[192 + t] = (t & 48) ? 1.f : 0.f;          // RS mask
  }
}

// ---------------------------------------------------------------------------
// Fused vertex-score + softmax + weight extraction.
// One wave per (b, i, 16-wide j tile). scoresT[v,j] = Veff(64x32) @ (q_i .* k_j)
// -> 4 WMMAs; softmax over v is 32 local values + one shfl_xor(16); then 4
// dot products against wvec planes give raw scalar/ap/bd/rs weights.
// ---------------------------------------------------------------------------
__global__ void scores_kernel(const float* __restrict__ qf, const float* __restrict__ kf,
                              const dbf16* __restrict__ Veff, const float* __restrict__ wvec,
                              float* __restrict__ Praw) {
  int jt = blockIdx.x, i = blockIdx.y, b = blockIdx.z;
  if (jt * 16 > i) return;                 // fully above the causal diagonal
  int lane = threadIdx.x;
  __shared__ float wsm[256];
  for (int t = lane; t < 256; t += 32) wsm[t] = wvec[t];
  __syncthreads();

  int c = lane & 15, h = lane >> 4;
  int j = jt * 16 + c;
  // B fragment: K=d (32), N=j (16);  e[t] = q[i, h*16+t] * k[j, h*16+t]
  const float* qrow = qf + ((size_t)b * S_LEN + i) * DV + h * 16;
  const float* krow = kf + ((size_t)b * S_LEN + j) * DV + h * 16;
  bf16x16 bfrag;
#pragma unroll
  for (int t = 0; t < 16; ++t) bfrag[t] = (dbf16)(qrow[t] * krow[t]);

  f32x8 cf[4];
#pragma unroll
  for (int t = 0; t < 4; ++t) {
    bf16x16 af = load_a_frag(Veff, DV, t * 16, 0, lane);
    f32x8 z;
#pragma unroll
    for (int r = 0; r < 8; ++r) z[r] = 0.f;
    cf[t] = __builtin_amdgcn_wmma_f32_16x16x32_bf16(false, af, false, bfrag,
                                                    (short)0, z, false, false);
  }
  // lane holds, for its column j, vertex values v = t*16 + h*8 + r
  float mx = -1e30f;
#pragma unroll
  for (int t = 0; t < 4; ++t)
#pragma unroll
    for (int r = 0; r < 8; ++r) mx = fmaxf(mx, cf[t][r]);
  mx = fmaxf(mx, __shfl_xor(mx, 16));
  float sum = 0.f;
#pragma unroll
  for (int t = 0; t < 4; ++t)
#pragma unroll
    for (int r = 0; r < 8; ++r) { float e = expf(cf[t][r] - mx); cf[t][r] = e; sum += e; }
  sum += __shfl_xor(sum, 16);
  float inv = 1.f / sum;

  float raw[4] = {0.f, 0.f, 0.f, 0.f};
#pragma unroll
  for (int t = 0; t < 4; ++t)
#pragma unroll
    for (int r = 0; r < 8; ++r) {
      float pv = cf[t][r] * inv;
      int v = t * 16 + h * 8 + r;
#pragma unroll
      for (int p = 0; p < 4; ++p) raw[p] += pv * wsm[p * 64 + v];
    }
#pragma unroll
  for (int p = 0; p < 4; ++p) raw[p] += __shfl_xor(raw[p], 16);

  if (h == 0 && j <= i) {
#pragma unroll
    for (int p = 0; p < 4; ++p)
      Praw[(((size_t)p * BATCH + b) * S_LEN + i) * S_LEN + j] = raw[p];
  }
}

// Row-normalize the 4 weight planes over j<=i (softmax-mask-renorm collapses
// to a single raw/(sum+eps)); emit bf16 for the combine GEMMs, zero above diag.
__global__ void rownorm_kernel(const float* __restrict__ Praw, dbf16* __restrict__ Pbf) {
  int i = blockIdx.x, b = blockIdx.y, p = blockIdx.z;
  const float* src = Praw + (((size_t)p * BATCH + b) * S_LEN + i) * S_LEN;
  dbf16* dst       = Pbf  + (((size_t)p * BATCH + b) * S_LEN + i) * S_LEN;
  int t = threadIdx.x;                     // 256
  __shared__ float red[256];
  float s = 0.f;
  for (int j = t; j < S_LEN; j += 256) if (j <= i) s += src[j];
  red[t] = s; __syncthreads();
  for (int off = 128; off > 0; off >>= 1) { if (t < off) red[t] += red[t + off]; __syncthreads(); }
  float inv = 1.f / (red[0] + 1e-8f);
  for (int j = t; j < S_LEN; j += 256)
    dst[j] = (j <= i) ? (dbf16)(src[j] * inv) : (dbf16)0.f;
}

// ---------------------------------------------------------------------------
// LayerNorm helpers / embed / residual
// ---------------------------------------------------------------------------
__device__ inline float block_ln(float v, float* red, float* stat, int t, int N) {
  red[t] = v; __syncthreads();
  if (t == 0) { float s = 0.f; for (int k = 0; k < N; ++k) s += red[k]; stat[0] = s / N; }
  __syncthreads();
  float d = v - stat[0];
  red[t] = d * d; __syncthreads();
  if (t == 0) { float s = 0.f; for (int k = 0; k < N; ++k) s += red[k]; stat[1] = s / N; }
  __syncthreads();
  return d * rsqrtf(stat[1] + 1e-5f);
}

__global__ void embed_kernel(const int* __restrict__ tokens,
                             const float* __restrict__ ea, const float* __restrict__ ef,
                             const float* __restrict__ eb,
                             const float* __restrict__ g, const float* __restrict__ bb,
                             float* __restrict__ x, dbf16* __restrict__ xbf) {
  __shared__ float red[DM]; __shared__ float stat[2];
  int row = blockIdx.x, t = threadIdx.x;   // blockDim = 192
  int s = row % S_LEN;
  int tok = tokens[row];
  float v;
  if (t < 64)       v = ea[tok * 64 + t];
  else if (t < 128) v = ef[tok * 64 + (t - 64)];
  else {
    int dp = t - 128;
    float div = expf((float)(dp & ~1) * (-logf(10000.f) / 64.f));
    float ang = (float)s * div;
    v = eb[tok * 64 + dp] + ((dp & 1) ? cosf(ang) : sinf(ang));
  }
  float y = block_ln(v, red, stat, t, DM) * g[t] + bb[t];
  size_t idx = (size_t)row * DM + t;
  x[idx] = y; xbf[idx] = (dbf16)y;
}

__global__ void ln_residual_kernel(float* __restrict__ x, dbf16* __restrict__ xbf,
                                   const float* __restrict__ src,
                                   const float* __restrict__ g, const float* __restrict__ bb) {
  __shared__ float red[DM]; __shared__ float stat[2];
  int row = blockIdx.x, t = threadIdx.x;   // blockDim = 192
  float v = src[(size_t)row * DM + t];
  float y = block_ln(v, red, stat, t, DM) * g[t] + bb[t];
  size_t idx = (size_t)row * DM + t;
  float xn = x[idx] + y;
  x[idx] = xn; xbf[idx] = (dbf16)xn;
}

__global__ void cvt_bf16_kernel(const float* __restrict__ src, dbf16* __restrict__ dst, int n) {
  int i = blockIdx.x * blockDim.x + threadIdx.x;
  if (i < n) dst[i] = (dbf16)src[i];
}

// ---------------------------------------------------------------------------
// Host orchestration
// ---------------------------------------------------------------------------
struct LayerIn {
  const float *Wq,*Wk,*vertex_emb,*vertex_gates,*Wv,*Wspec,*Wap,*Wbd,*Wres,
              *Wout,*bout,*ln1_g,*ln1_b,*W1,*b1,*W2,*b2,*ln2_g,*ln2_b;
};

extern "C" void kernel_launch(void* const* d_in, const int* in_sizes, int n_in,
                              void* d_out, int out_size, void* d_ws, size_t ws_size,
                              hipStream_t stream) {
  (void)n_in; (void)out_size; (void)ws_size;
  auto f = [&](int i) { return (const float*)d_in[i]; };

  const float *embed_a,*embed_f,*embed_b,*emb_g,*emb_b,*lm_w,*lm_b;
  const int* tokens;
  LayerIn L[2];

  if (in_sizes[0] == NVOC * D3) {          // dict insertion order
    embed_a = f(0); embed_f = f(1); embed_b = f(2); emb_g = f(3); emb_b = f(4);
    for (int l = 0; l < 2; ++l) {
      int b0 = 5 + 19 * l;
      L[l] = { f(b0+0), f(b0+1), f(b0+2), f(b0+3), f(b0+4), f(b0+5), f(b0+6),
               f(b0+7), f(b0+8), f(b0+9), f(b0+10), f(b0+11), f(b0+12),
               f(b0+13), f(b0+14), f(b0+15), f(b0+16), f(b0+17), f(b0+18) };
    }
    lm_w = f(43); lm_b = f(44); tokens = (const int*)d_in[45];
  } else {                                 // jax pytree (sorted keys) order
    emb_b = f(0); emb_g = f(1); embed_a = f(2); embed_b = f(3); embed_f = f(4);
    for (int l = 0; l < 2; ++l) {
      int b0 = 5 + 19 * l;
      L[l].W1 = f(b0+0);  L[l].W2 = f(b0+1);  L[l].Wap = f(b0+2); L[l].Wbd = f(b0+3);
      L[l].Wk = f(b0+4);  L[l].Wout = f(b0+5); L[l].Wq = f(b0+6); L[l].Wres = f(b0+7);
      L[l].Wspec = f(b0+8); L[l].Wv = f(b0+9); L[l].b1 = f(b0+10); L[l].b2 = f(b0+11);
      L[l].bout = f(b0+12); L[l].ln1_b = f(b0+13); L[l].ln1_g = f(b0+14);
      L[l].ln2_b = f(b0+15); L[l].ln2_g = f(b0+16);
      L[l].vertex_emb = f(b0+17); L[l].vertex_gates = f(b0+18);
    }
    lm_b = f(43); lm_w = f(44); tokens = (const int*)d_in[45];
  }

  // ---- workspace bump allocator ----
  char* base = (char*)d_ws; size_t off = 0;
  auto alloc = [&](size_t bytes) { off = (off + 255) & ~(size_t)255;
                                   void* p = base + off; off += bytes; return p; };
  auto af32 = [&](size_t n) { return (float*)alloc(n * 4); };
  auto abf  = [&](size_t n) { return (dbf16*)alloc(n * 2); };

  float* x_f    = af32(MROWS * DM);   dbf16* x_bf   = abf(MROWS * DM);
  float* qf     = af32(MROWS * DV);   float* kf     = af32(MROWS * DV);
  float* xv_f   = af32(MROWS * DM);   dbf16* xvT    = abf((size_t)DM * MROWS);  // (192,1024)
  float* xap_f  = af32(MROWS * D3);   dbf16* xapT   = abf((size_t)D3 * MROWS);  // (64,1024)
  float* xbd_f  = af32(MROWS * D3);   dbf16* xbdT   = abf((size_t)D3 * MROWS);
  float* xrs_f  = af32(MROWS * D3);   dbf16* xrsT   = abf((size_t)D3 * MROWS);
  float* Praw   = af32((size_t)4 * BATCH * S_LEN * S_LEN);
  dbf16* Pbf    = abf((size_t)4 * BATCH * S_LEN * S_LEN);
  float* comb_f = af32(MROWS * DM);   dbf16* comb_bf = abf(MROWS * DM);
  float* res_f  = af32(MROWS * DM);
  float* h_f    = af32(MROWS * DFF);  dbf16* h_bf   = abf(MROWS * DFF);
  float* y2_f   = af32(MROWS * DM);
  dbf16* veff   = abf(64 * DV);       float* wvec   = af32(4 * 64);
  dbf16 *wq_b = abf(DV*DM), *wk_b = abf(DV*DM), *wv_b = abf(DM*DM);
  dbf16 *wap_b = abf(D3*DM), *wbd_b = abf(D3*DM), *wrs_b = abf(D3*DM);
  dbf16 *wout_b = abf(DM*DM), *w1_b = abf(DFF*DM), *w2_b = abf(DM*DFF);
  dbf16 *lmw_b = abf(NVOC*DM);

  auto CVT = [&](const float* s, dbf16* d, int n) {
    cvt_bf16_kernel<<<(n + 255) / 256, 256, 0, stream>>>(s, d, n);
  };
  auto TRN = [&](const float* s, dbf16* d, int R, int C) {
    transpose_bf16_kernel<<<dim3(R / 32, C / 32), dim3(32, 8), 0, stream>>>(s, d, R, C);
  };
  auto G = [&](const dbf16* A, int lda, const dbf16* Bp, int ldb,
               const float* Cin, float* D, dbf16* Dbf, int ldd, const float* bias,
               int M, int N, int K, int causal, int act) {
    gemm_wmma_kernel<<<dim3(M / 32, N / 32), 32, 0, stream>>>(
        A, lda, Bp, ldb, Cin, D, Dbf, ldd, bias, K, causal, act);
  };

  // ---- embed + LN ----
  embed_kernel<<<MROWS, DM, 0, stream>>>(tokens, embed_a, embed_f, embed_b,
                                         emb_g, emb_b, x_f, x_bf);

  for (int l = 0; l < 2; ++l) {
    const LayerIn& p = L[l];
    CVT(p.Wq, wq_b, DV*DM);   CVT(p.Wk, wk_b, DV*DM);   CVT(p.Wv, wv_b, DM*DM);
    CVT(p.Wap, wap_b, D3*DM); CVT(p.Wbd, wbd_b, D3*DM); CVT(p.Wres, wrs_b, D3*DM);
    CVT(p.Wout, wout_b, DM*DM); CVT(p.W1, w1_b, DFF*DM); CVT(p.W2, w2_b, DM*DFF);
    prep_layer_kernel<<<1, 512, 0, stream>>>(p.vertex_emb, p.Wspec, p.vertex_gates,
                                             veff, wvec);
    // projections (weights already (N,K) -> contiguous B loads)
    G(x_bf, DM, wq_b, DM, nullptr, qf, nullptr, DV, nullptr, MROWS, DV, DM, 0, 0);
    G(x_bf, DM, wk_b, DM, nullptr, kf, nullptr, DV, nullptr, MROWS, DV, DM, 0, 0);
    G(x_bf, DM, wv_b, DM, nullptr, xv_f, nullptr, DM, nullptr, MROWS, DM, DM, 0, 0);
    G(x_bf, DM, wap_b, DM, nullptr, xap_f, nullptr, D3, nullptr, MROWS, D3, DM, 0, 0);
    G(x_bf, DM, wbd_b, DM, nullptr, xbd_f, nullptr, D3, nullptr, MROWS, D3, DM, 0, 0);
    G(x_bf, DM, wrs_b, DM, nullptr, xrs_f, nullptr, D3, nullptr, MROWS, D3, DM, 0, 0);
    // transpose value streams to (channel, row) bf16 for contiguous B-frags
    TRN(xv_f,  xvT,  MROWS, DM);
    TRN(xap_f, xapT, MROWS, D3);
    TRN(xbd_f, xbdT, MROWS, D3);
    TRN(xrs_f, xrsT, MROWS, D3);
    // fused vertex scores -> softmax -> 4 raw weight planes
    scores_kernel<<<dim3(S_LEN / 16, S_LEN, BATCH), 32, 0, stream>>>(qf, kf, veff, wvec, Praw);
    rownorm_kernel<<<dim3(S_LEN, BATCH, 4), 256, 0, stream>>>(Praw, Pbf);
    // attention combine: per batch, causal (S,S)@(S,*) GEMMs, all-contiguous B
    for (int b = 0; b < BATCH; ++b) {
      const size_t PS = (size_t)S_LEN * S_LEN;
      const dbf16* P0 = Pbf + (0 * BATCH + b) * PS;
      const dbf16* P1 = Pbf + (1 * BATCH + b) * PS;
      const dbf16* P2 = Pbf + (2 * BATCH + b) * PS;
      const dbf16* P3 = Pbf + (3 * BATCH + b) * PS;
      float* cb = comb_f + (size_t)b * S_LEN * DM;
      G(P0, S_LEN, xvT  + (size_t)b * S_LEN, MROWS, nullptr, cb, nullptr, DM, nullptr, S_LEN, DM, S_LEN, 1, 0);
      G(P1, S_LEN, xapT + (size_t)b * S_LEN, MROWS, cb,       cb,       nullptr, DM, nullptr, S_LEN, D3, S_LEN, 1, 0);
      G(P2, S_LEN, xbdT + (size_t)b * S_LEN, MROWS, cb + 64,  cb + 64,  nullptr, DM, nullptr, S_LEN, D3, S_LEN, 1, 0);
      G(P3, S_LEN, xrsT + (size_t)b * S_LEN, MROWS, cb + 128, cb + 128, nullptr, DM, nullptr, S_LEN, D3, S_LEN, 1, 0);
    }
    CVT(comb_f, comb_bf, MROWS * DM);
    // output proj + LN + residual
    G(comb_bf, DM, wout_b, DM, nullptr, res_f, nullptr, DM, p.bout, MROWS, DM, DM, 0, 0);
    ln_residual_kernel<<<MROWS, DM, 0, stream>>>(x_f, x_bf, res_f, p.ln1_g, p.ln1_b);
    // FFN
    G(x_bf, DM, w1_b, DM, nullptr, h_f, h_bf, DFF, p.b1, MROWS, DFF, DM, 0, 1);
    G(h_bf, DFF, w2_b, DFF, nullptr, y2_f, nullptr, DM, p.b2, MROWS, DM, DFF, 0, 0);
    ln_residual_kernel<<<MROWS, DM, 0, stream>>>(x_f, x_bf, y2_f, p.ln2_g, p.ln2_b);
  }

  // ---- LM head ----
  CVT(lm_w, lmw_b, NVOC * DM);
  G(x_bf, DM, lmw_b, DM, nullptr, (float*)d_out, nullptr, NVOC, lm_b,
    MROWS, NVOC, DM, 0, 0);
}
